// WindowAttention_54614804136405
// MI455X (gfx1250) — compile-verified, matrix operand order checked by reading, not run on hardware
//
#include <hip/hip_runtime.h>

// ---------------------------------------------------------------------------
// Focal window attention for MI455X (gfx1250, wave32, WMMA bf16 16x16x32)
// B=4 H=W=112 C=256 WS=7 WA=49 NWIN=256 HEADS=8 HD=32 L_ROLL=181 L=230 Lp=256
// ---------------------------------------------------------------------------

typedef __attribute__((ext_vector_type(16))) __bf16 v16bf;
typedef __attribute__((ext_vector_type(8)))  float  v8f;

union FragBF {
    v16bf          v;
    unsigned short us[16];
    uint4          q[2];
};

union Pack8 {
    unsigned short us[8];
    uint4          q;
};

// Native bf16 convert (lowers to v_cvt hardware op; pairs fuse to pk form).
__device__ inline unsigned short f2bf(float f) {
    return __builtin_bit_cast(unsigned short, (__bf16)f);
}

__device__ inline v8f wmma_bf16(const FragBF& a, const FragBF& b, v8f c) {
    return __builtin_amdgcn_wmma_f32_16x16x32_bf16(
        false, a.v, false, b.v, (short)0, c, false, false);
}

__device__ inline v8f zero8() {
    v8f z;
#pragma unroll
    for (int i = 0; i < 8; ++i) z[i] = 0.0f;
    return z;
}

// A/B fragment from packed bf16 memory (16B-aligned).
// p points at row start + kbase; elems 0..7 <- p[0..7], 8..15 <- p[16..23].
__device__ inline void load_frag_bf(FragBF& f, const unsigned short* __restrict__ p) {
    f.q[0] = *(const uint4*)(p);
    f.q[1] = *(const uint4*)(p + 16);
}

// token (b*12544 + h*112 + w) -> (window, position-in-window)
__device__ inline void tok2wp(int tok, int& win, int& pos) {
    int b = tok / 12544, rem = tok - b * 12544;
    int h = rem / 112, w = rem - h * 112;
    win = b * 256 + (h / 7) * 16 + (w / 7);
    pos = (h % 7) * 7 + (w % 7);
}

// j in [0,132) -> flat index into the 4x7x7 rolled mask stack (VALID_IND)
__device__ inline int valid_ind(int j) {
    int m = j / 33, jj = j - m * 33, r, c;
    if (m == 0)      { if (jj < 12) { r = jj / 3;  c = 4 + jj % 3; }
                       else { int t = jj - 12; r = 4 + t / 7; c = t % 7; } }
    else if (m == 1) { if (jj < 12) { r = jj / 3;  c = jj % 3; }
                       else { int t = jj - 12; r = 4 + t / 7; c = t % 7; } }
    else if (m == 2) { if (jj < 21) { r = jj / 7;  c = jj % 7; }
                       else { int t = jj - 21; r = 3 + t / 3; c = 4 + t % 3; } }
    else             { if (jj < 21) { r = jj / 7;  c = jj % 7; }
                       else { int t = jj - 21; r = 3 + t / 3; c = t % 3; } }
    return m * 49 + r * 7 + c;
}

// ---------------------------------------------------------------------------
__global__ void __launch_bounds__(256) zero_kernel(uint4* p, long n) {
    long i = blockIdx.x * (long)blockDim.x + threadIdx.x;
    long s = gridDim.x * (long)blockDim.x;
    uint4 z = make_uint4(0u, 0u, 0u, 0u);
    for (; i < n; i += s) p[i] = z;
}

// Elementwise fp32 -> bf16, 8 elements per thread-iteration.
__global__ void __launch_bounds__(256) cvt_f32_bf16(
    const float* __restrict__ src, unsigned short* __restrict__ dst, long n8) {
    long i = blockIdx.x * (long)blockDim.x + threadIdx.x;
    long s = gridDim.x * (long)blockDim.x;
    for (; i < n8; i += s) {
        const float4 a = ((const float4*)src)[i * 2];
        const float4 b = ((const float4*)src)[i * 2 + 1];
        Pack8 o;
        o.us[0] = f2bf(a.x); o.us[1] = f2bf(a.y);
        o.us[2] = f2bf(a.z); o.us[3] = f2bf(a.w);
        o.us[4] = f2bf(b.x); o.us[5] = f2bf(b.y);
        o.us[6] = f2bf(b.z); o.us[7] = f2bf(b.w);
        ((uint4*)dst)[i] = o.q;
    }
}

// Convert the 4 weight matrices (q,k,v,proj; each 256x256) into wbf[4][256][256].
__global__ void __launch_bounds__(256) cvt_weights(
    const float* __restrict__ qw, const float* __restrict__ kw,
    const float* __restrict__ vw, const float* __restrict__ pw,
    unsigned short* __restrict__ wbf) {
    const int i = blockIdx.x * blockDim.x + threadIdx.x;  // chunk of 8 floats
    if (i >= 32768) return;
    const int w = i >> 13, off8 = (i & 8191);
    const float* src = (w == 0) ? qw : (w == 1) ? kw : (w == 2) ? vw : pw;
    const float4 a = ((const float4*)src)[off8 * 2];
    const float4 b = ((const float4*)src)[off8 * 2 + 1];
    Pack8 o;
    o.us[0] = f2bf(a.x); o.us[1] = f2bf(a.y);
    o.us[2] = f2bf(a.z); o.us[3] = f2bf(a.w);
    o.us[4] = f2bf(b.x); o.us[5] = f2bf(b.y);
    o.us[6] = f2bf(b.z); o.us[7] = f2bf(b.w);
    ((uint4*)(wbf + w * 65536))[off8] = o.q;
}

// Fused gather (VALID_IND) + convert of x_expanded -> xebf (135168 x 256 bf16).
__global__ void __launch_bounds__(256) gather_cvt_xe(
    const float* __restrict__ xe, unsigned short* __restrict__ xebf) {
    const long total = 135168L * 32;  // chunks of 8 elements
    long i = blockIdx.x * (long)blockDim.x + threadIdx.x;
    long s = gridDim.x * (long)blockDim.x;
    for (; i < total; i += s) {
        const int row = (int)(i >> 5), c8 = (int)(i & 31);
        const int win = row / 132, j = row - win * 132;
        const long src = ((long)(win * 196 + valid_ind(j))) * 256 + c8 * 8;
        const float4 a = *(const float4*)(xe + src);
        const float4 b = *(const float4*)(xe + src + 4);
        Pack8 o;
        o.us[0] = f2bf(a.x); o.us[1] = f2bf(a.y);
        o.us[2] = f2bf(a.z); o.us[3] = f2bf(a.w);
        o.us[4] = f2bf(b.x); o.us[5] = f2bf(b.y);
        o.us[6] = f2bf(b.z); o.us[7] = f2bf(b.w);
        ((uint4*)(xebf + (long)row * 256))[c8] = o.q;
    }
}

// ---------------------------------------------------------------------------
// QKV projection of xbf (50176 x 256 bf16) with window scatter.
// grid 784, block 256 (8 waves). Each block: 64 tokens x 768 output cols.
__global__ void __launch_bounds__(256) qkv_proj_x(
    const unsigned short* __restrict__ xbf,
    const unsigned short* __restrict__ wbf,   // [4][256][256], 0=q 1=k 2=v
    const float* __restrict__ qb, const float* __restrict__ kb,
    const float* __restrict__ vb,
    unsigned short* __restrict__ qbuf,   // [1024][8][64][32]
    unsigned short* __restrict__ kbuf,   // [1024][8][256][32]
    unsigned short* __restrict__ vbufT)  // [1024][8][32][256]
{
    const int tid = threadIdx.x, lane = tid & 31, wv = tid >> 5;
    const int lo = lane & 15, hi = lane >> 4;
    const int rowbase = blockIdx.x * 64;

    for (int ntl = 0; ntl < 6; ++ntl) {
        const int nt = wv * 6 + ntl;                 // 0..47
        const int kind = nt >> 4;                    // 0=q 1=k 2=v
        const float* Bb = (kind == 0) ? qb : (kind == 1) ? kb : vb;
        const int ncol = (nt & 15) * 16 + lo;        // 0..255
        const unsigned short* W = wbf + kind * 65536 + ncol * 256;

        v8f acc[4];
#pragma unroll
        for (int mt = 0; mt < 4; ++mt) acc[mt] = zero8();

        for (int kc = 0; kc < 8; ++kc) {
            const int kbase = kc * 32 + hi * 8;
            FragBF bf; load_frag_bf(bf, W + kbase);
#pragma unroll
            for (int mt = 0; mt < 4; ++mt) {
                const int row = rowbase + mt * 16 + lo;
                FragBF af; load_frag_bf(af, xbf + row * 256 + kbase);
                acc[mt] = wmma_bf16(af, bf, acc[mt]);
            }
        }
        const float bias = Bb[ncol];
        const int head = ncol >> 5, d = ncol & 31;
#pragma unroll
        for (int mt = 0; mt < 4; ++mt) {
#pragma unroll
            for (int r = 0; r < 8; ++r) {
                const int tok = rowbase + mt * 16 + r + 8 * hi;
                int win, pos; tok2wp(tok, win, pos);
                float val = acc[mt][r] + bias;
                if (kind == 0) {
                    val *= 0.17677669529663687f;     // 32^-0.5
                    qbuf[((win * 8 + head) * 64 + pos) * 32 + d] = f2bf(val);
                } else if (kind == 1) {
                    kbuf[((win * 8 + head) * 256 + pos) * 32 + d] = f2bf(val);
                } else {
                    vbufT[((win * 8 + head) * 32 + d) * 256 + pos] = f2bf(val);
                }
            }
        }
    }
}

// ---------------------------------------------------------------------------
// K/V projection of pre-gathered xebf rows -> positions 49..180.
// rows = 1024*132 = 135168, grid 2112.
__global__ void __launch_bounds__(256) kv_proj_exp(
    const unsigned short* __restrict__ xebf,         // [135168][256]
    const unsigned short* __restrict__ wbf,
    const float* __restrict__ kb, const float* __restrict__ vb,
    unsigned short* __restrict__ kbuf, unsigned short* __restrict__ vbufT)
{
    const int tid = threadIdx.x, lane = tid & 31, wv = tid >> 5;
    const int lo = lane & 15, hi = lane >> 4;
    const int rowbase = blockIdx.x * 64;

    for (int ntl = 0; ntl < 4; ++ntl) {
        const int nt = wv * 4 + ntl;                 // 0..31
        const bool is_k = nt < 16;
        const unsigned short* W = wbf + (is_k ? 1 : 2) * 65536;
        const float* Bb = is_k ? kb : vb;
        const int ncol = (nt & 15) * 16 + lo;

        v8f acc[4];
#pragma unroll
        for (int mt = 0; mt < 4; ++mt) acc[mt] = zero8();

        for (int kc = 0; kc < 8; ++kc) {
            const int kbase = kc * 32 + hi * 8;
            FragBF bf; load_frag_bf(bf, W + ncol * 256 + kbase);
#pragma unroll
            for (int mt = 0; mt < 4; ++mt) {
                const int gr = rowbase + mt * 16 + lo;
                FragBF af; load_frag_bf(af, xebf + (long)gr * 256 + kbase);
                acc[mt] = wmma_bf16(af, bf, acc[mt]);
            }
        }
        const float bias = Bb[ncol];
        const int head = ncol >> 5, d = ncol & 31;
#pragma unroll
        for (int mt = 0; mt < 4; ++mt) {
#pragma unroll
            for (int r = 0; r < 8; ++r) {
                const int gr = rowbase + mt * 16 + r + 8 * hi;
                const int win = gr / 132, j = gr - win * 132;
                const int pos = 49 + j;
                const float val = acc[mt][r] + bias;
                if (is_k)
                    kbuf[((win * 8 + head) * 256 + pos) * 32 + d] = f2bf(val);
                else
                    vbufT[((win * 8 + head) * 32 + d) * 256 + pos] = f2bf(val);
            }
        }
    }
}

// ---------------------------------------------------------------------------
// K/V projection of xpbf -> positions 181..229. rows = 50176, grid 784.
__global__ void __launch_bounds__(256) kv_proj_pool(
    const unsigned short* __restrict__ xpbf,         // [50176][256]
    const unsigned short* __restrict__ wbf,
    const float* __restrict__ kb, const float* __restrict__ vb,
    unsigned short* __restrict__ kbuf, unsigned short* __restrict__ vbufT)
{
    const int tid = threadIdx.x, lane = tid & 31, wv = tid >> 5;
    const int lo = lane & 15, hi = lane >> 4;
    const int rowbase = blockIdx.x * 64;

    for (int ntl = 0; ntl < 4; ++ntl) {
        const int nt = wv * 4 + ntl;
        const bool is_k = nt < 16;
        const unsigned short* W = wbf + (is_k ? 1 : 2) * 65536;
        const float* Bb = is_k ? kb : vb;
        const int ncol = (nt & 15) * 16 + lo;

        v8f acc[4];
#pragma unroll
        for (int mt = 0; mt < 4; ++mt) acc[mt] = zero8();

        for (int kc = 0; kc < 8; ++kc) {
            const int kbase = kc * 32 + hi * 8;
            FragBF bf; load_frag_bf(bf, W + ncol * 256 + kbase);
#pragma unroll
            for (int mt = 0; mt < 4; ++mt) {
                const int gr = rowbase + mt * 16 + lo;
                FragBF af; load_frag_bf(af, xpbf + gr * 256 + kbase);
                acc[mt] = wmma_bf16(af, bf, acc[mt]);
            }
        }
        const float bias = Bb[ncol];
        const int head = ncol >> 5, d = ncol & 31;
#pragma unroll
        for (int mt = 0; mt < 4; ++mt) {
#pragma unroll
            for (int r = 0; r < 8; ++r) {
                const int gr = rowbase + mt * 16 + r + 8 * hi;
                const int win = gr / 49, p = gr - win * 49;
                const int pos = 181 + p;
                const float val = acc[mt][r] + bias;
                if (is_k)
                    kbuf[((win * 8 + head) * 256 + pos) * 32 + d] = f2bf(val);
                else
                    vbufT[((win * 8 + head) * 32 + d) * 256 + pos] = f2bf(val);
            }
        }
    }
}

// ---------------------------------------------------------------------------
// Attention per (window, head). grid 8192, block 256 (8 waves).
#define SSTR 264   // f32 score row stride
#define ASTR 264   // bf16 attn row stride (multiple of 8 for 16B loads)

__global__ void __launch_bounds__(256) attn_kernel(
    const unsigned short* __restrict__ qbuf,
    const unsigned short* __restrict__ kbuf,
    const unsigned short* __restrict__ vbufT,
    const float* __restrict__ rpb,     // (8,49,181)
    const float* __restrict__ rpbp,    // (8,49,49)
    const float* __restrict__ maskp,   // (256,49)
    unsigned short* __restrict__ ctx)  // (50176,256) bf16
{
    extern __shared__ char smem[];
    float*          sc = (float*)smem;                              // [64][SSTR]
    unsigned short* ab = (unsigned short*)(smem + 64 * SSTR * 4);   // [64][ASTR]

    const int tid = threadIdx.x, lane = tid & 31, wv = tid >> 5;
    const int lo = lane & 15, hi = lane >> 4;
    const int win = blockIdx.x >> 3, head = blockIdx.x & 7;

    const unsigned short* qp = qbuf  + (win * 8 + head) * 64 * 32;
    const unsigned short* kp = kbuf  + (win * 8 + head) * 256 * 32;
    const unsigned short* vp = vbufT + (win * 8 + head) * 32 * 256;

    // ---- Phase 1: scores(64x256) = q @ k^T, 64 WMMA tiles, 8 per wave ----
#pragma unroll
    for (int t = 0; t < 8; ++t) {
        const int tile = wv * 8 + t;
        const int mt = tile >> 4, nt = tile & 15;
        const int kbase = hi * 8;
        FragBF a; load_frag_bf(a, qp + (mt * 16 + lo) * 32 + kbase);
        FragBF b; load_frag_bf(b, kp + (nt * 16 + lo) * 32 + kbase);
        v8f c = zero8();
        c = wmma_bf16(a, b, c);
        const int m = mt * 16 + 8 * hi, n = nt * 16 + lo;
#pragma unroll
        for (int r = 0; r < 8; ++r) sc[(m + r) * SSTR + n] = c[r];
    }
    __syncthreads();

    // ---- Phase 2: bias add + softmax; 4 threads per row ----
    {
        const int r = tid >> 2, sub = tid & 3;
        unsigned short* arow = ab + r * ASTR;
        if (r < 49) {
            const float* rpr = rpb  + (head * 49 + r) * 181;
            const float* rpp = rpbp + (head * 49 + r) * 49;
            const float* mkr = maskp + (win & 255) * 49;
            float* srow = sc + r * SSTR;
            float mx = -1e30f;
            for (int i = 0; i < 64; ++i) {
                const int c = sub * 64 + i;
                float v = srow[c];
                if (c < 181)      v += rpr[c];
                else if (c < 230) v += rpp[c - 181] + mkr[c - 181];
                else              v  = -1e30f;
                srow[c] = v;
                mx = fmaxf(mx, v);
            }
            mx = fmaxf(mx, __shfl_xor(mx, 1, 4));
            mx = fmaxf(mx, __shfl_xor(mx, 2, 4));
            float sum = 0.0f;
            for (int i = 0; i < 64; ++i) {
                const int c = sub * 64 + i;
                const float e = __expf(srow[c] - mx);
                srow[c] = e;
                sum += e;
            }
            sum += __shfl_xor(sum, 1, 4);
            sum += __shfl_xor(sum, 2, 4);
            const float inv = 1.0f / sum;
            for (int i = 0; i < 64; ++i) {
                const int c = sub * 64 + i;
                arow[c] = (c < 230) ? f2bf(srow[c] * inv) : (unsigned short)0;
            }
        } else {
            for (int i = 0; i < 64; ++i) arow[sub * 64 + i] = 0;
        }
    }
    __syncthreads();

    // ---- Phase 3: out(64x32) = attn @ v; one 16x16 tile per wave ----
    {
        const int mt = wv >> 1, ntd = wv & 1;
        v8f acc = zero8();
        for (int kc = 0; kc < 8; ++kc) {
            const int kbase = kc * 32 + hi * 8;
            FragBF a; load_frag_bf(a, ab + (mt * 16 + lo) * ASTR + kbase);
            FragBF b; load_frag_bf(b, vp + (ntd * 16 + lo) * 256 + kbase);
            acc = wmma_bf16(a, b, acc);
        }
        const int dcol = head * 32 + ntd * 16 + lo;
#pragma unroll
        for (int r = 0; r < 8; ++r) {
            const int m = mt * 16 + r + 8 * hi;
            if (m < 49) ctx[(win * 49 + m) * 256 + dcol] = f2bf(acc[r]);
        }
    }
}

// ---------------------------------------------------------------------------
// Output projection: out(50176x256 f32) = ctx(bf16) @ proj_w^T + proj_b
__global__ void __launch_bounds__(256) proj_kernel(
    const unsigned short* __restrict__ ctx,
    const unsigned short* __restrict__ wbf,  // slot 3 = proj_w
    const float* __restrict__ pb,
    float* __restrict__ out)
{
    const int tid = threadIdx.x, lane = tid & 31, wv = tid >> 5;
    const int lo = lane & 15, hi = lane >> 4;
    const int rowbase = blockIdx.x * 64;
    const unsigned short* W = wbf + 3 * 65536;

    for (int ntl = 0; ntl < 2; ++ntl) {
        const int nt = wv * 2 + ntl;
        const int ncol = nt * 16 + lo;

        v8f acc[4];
#pragma unroll
        for (int mt = 0; mt < 4; ++mt) acc[mt] = zero8();

        for (int kc = 0; kc < 8; ++kc) {
            const int kbase = kc * 32 + hi * 8;
            FragBF bf; load_frag_bf(bf, W + ncol * 256 + kbase);
#pragma unroll
            for (int mt = 0; mt < 4; ++mt) {
                FragBF af;
                load_frag_bf(af, ctx + (rowbase + mt * 16 + lo) * 256 + kbase);
                acc[mt] = wmma_bf16(af, bf, acc[mt]);
            }
        }
        const float bias = pb[ncol];
#pragma unroll
        for (int mt = 0; mt < 4; ++mt) {
#pragma unroll
            for (int r = 0; r < 8; ++r) {
                const int row = rowbase + mt * 16 + r + 8 * hi;
                out[row * 256 + ncol] = acc[mt][r] + bias;
            }
        }
    }
}

// ---------------------------------------------------------------------------
extern "C" void kernel_launch(void* const* d_in, const int* in_sizes, int n_in,
                              void* d_out, int out_size, void* d_ws, size_t ws_size,
                              hipStream_t stream) {
    const float* x   = (const float*)d_in[0];
    const float* xe  = (const float*)d_in[1];
    const float* xp  = (const float*)d_in[2];
    const float* mp  = (const float*)d_in[3];
    const float* qw  = (const float*)d_in[4];
    const float* qb  = (const float*)d_in[5];
    const float* kw  = (const float*)d_in[6];
    const float* kb  = (const float*)d_in[7];
    const float* vw  = (const float*)d_in[8];
    const float* vb  = (const float*)d_in[9];
    const float* pw  = (const float*)d_in[10];
    const float* pb  = (const float*)d_in[11];
    const float* rpb = (const float*)d_in[12];
    const float* rpp = (const float*)d_in[13];

    char* ws = (char*)d_ws;
    unsigned short* qbuf  = (unsigned short*)(ws);                 // 32 MB
    unsigned short* vbufT = (unsigned short*)(ws + 33554432L);     // 128 MB
    unsigned short* kbuf  = (unsigned short*)(ws + 167772160L);    // 128 MB
    unsigned short* xbf   = (unsigned short*)(ws + 301989888L);    // 25.7 MB
    unsigned short* ctx   = xbf;          // reuses xbf region (xbf dead by then)
    unsigned short* xebf  = (unsigned short*)(ws + 327680000L);    // 69.2 MB
    unsigned short* xpbf  = (unsigned short*)(ws + 396886016L);    // 25.7 MB
    unsigned short* wbf   = (unsigned short*)(ws + 422576128L);    // 0.5 MB
    float* out = (float*)d_out;

    // Zero q pad rows + v^T pad cols (contiguous first 160 MB region).
    zero_kernel<<<4096, 256, 0, stream>>>((uint4*)ws, 10485760L);

    // Pre-convert operands to bf16 (and gather x_expanded[VALID_IND]).
    cvt_f32_bf16<<<2048, 256, 0, stream>>>(x,  xbf,  1605632L);
    cvt_f32_bf16<<<2048, 256, 0, stream>>>(xp, xpbf, 1605632L);
    cvt_weights <<<128,  256, 0, stream>>>(qw, kw, vw, pw, wbf);
    gather_cvt_xe<<<4096, 256, 0, stream>>>(xe, xebf);

    // Projections (all-WMMA inner loops).
    qkv_proj_x <<<784,  256, 0, stream>>>(xbf,  wbf, qb, kb, vb,
                                          qbuf, kbuf, vbufT);
    kv_proj_exp<<<2112, 256, 0, stream>>>(xebf, wbf, kb, vb, kbuf, vbufT);
    kv_proj_pool<<<784, 256, 0, stream>>>(xpbf, wbf, kb, vb, kbuf, vbufT);

    const int lds_bytes = 64 * SSTR * 4 + 64 * ASTR * 2;  // 101376 B
    hipFuncSetAttribute((const void*)attn_kernel,
                        hipFuncAttributeMaxDynamicSharedMemorySize, lds_bytes);
    attn_kernel<<<8192, 256, lds_bytes, stream>>>(qbuf, kbuf, vbufT,
                                                  rpb, rpp, mp, ctx);

    proj_kernel<<<784, 256, 0, stream>>>(ctx, wbf, pb, out);
}
